// SpikingMoEFFN_1563368095962
// MI455X (gfx1250) — compile-verified, matrix operand order checked by reading, not run on hardware
//
#include <hip/hip_runtime.h>
#include <hip/hip_bf16.h>
#include <cstddef>

// ---------------- CDNA5 WMMA types ----------------
typedef __attribute__((ext_vector_type(16))) __bf16 v16bf;
typedef __attribute__((ext_vector_type(8)))  __bf16 v8bf;
typedef __attribute__((ext_vector_type(8)))  float  v8f;

#define D_MODEL 1024
#define HIDDEN  2816
#define N_EXP   8
#define T_TOK   2048
#define T_TILE  16
#define XPAD    (D_MODEL + 8)   // padded bf16 row stride in LDS
#define HPAD    (HIDDEN + 8)

// ============================================================
// Kernel 1: spiking gate -> combine weights [T, 8]
// ============================================================
__global__ __launch_bounds__(256) void gate_kernel(
    const float* __restrict__ x, const float* __restrict__ Wg,
    const float* __restrict__ bg, float* __restrict__ combine)
{
    int t    = blockIdx.x * 8 + (threadIdx.x >> 5);
    int lane = threadIdx.x & 31;
    if (t >= T_TOK) return;

    float acc[N_EXP];
#pragma unroll
    for (int e = 0; e < N_EXP; ++e) acc[e] = 0.0f;

    const float* xr = x + (size_t)t * D_MODEL;
    for (int k = lane; k < D_MODEL; k += 32) {
        float xv = xr[k];
        const float* w = Wg + (size_t)k * N_EXP;
#pragma unroll
        for (int e = 0; e < N_EXP; ++e) acc[e] += xv * w[e];
    }
#pragma unroll
    for (int off = 16; off > 0; off >>= 1) {
#pragma unroll
        for (int e = 0; e < N_EXP; ++e)
            acc[e] += __shfl_down(acc[e], off, 32);
    }
    if (lane == 0) {
        int spike[N_EXP];
        int cnt = 0;
        float w_[N_EXP];
#pragma unroll
        for (int e = 0; e < N_EXP; ++e) {
            float lg = acc[e] + bg[e];
            spike[e] = (lg > 1.0f) ? 1 : 0;
            cnt += spike[e];
            w_[e] = 0.0f;
        }
        if (cnt >= 2) {
            int found = 0;
#pragma unroll
            for (int e = 0; e < N_EXP; ++e)
                if (spike[e] && found < 2) { w_[e] = 0.5f; ++found; }
        } else if (cnt == 1) {
            int j = 0;
#pragma unroll
            for (int e = 0; e < N_EXP; ++e) if (spike[e]) j = e;
            int o = (j == 0) ? 1 : 0;
            w_[j] = 0.73105857863f;   // softmax(1,0)
            w_[o] = 0.26894142137f;
        } else {
            w_[0] = 0.5f; w_[1] = 0.5f;
        }
        float* co = combine + (size_t)t * N_EXP;
#pragma unroll
        for (int e = 0; e < N_EXP; ++e) co[e] = w_[e];
    }
}

// ============================================================
// Kernel 2: x fp32 -> bf16 (flat)
// ============================================================
__global__ __launch_bounds__(256) void cvt_x_kernel(
    const float4* __restrict__ x, __bf16* __restrict__ xb)
{
    int i = blockIdx.x * 256 + threadIdx.x;     // 4 floats per thread
    float4 v = x[i];
    __bf16* d = xb + (size_t)i * 4;
    d[0] = (__bf16)v.x; d[1] = (__bf16)v.y;
    d[2] = (__bf16)v.z; d[3] = (__bf16)v.w;
}

// ============================================================
// Kernel 3: weight fp32 [E][R][C] -> transposed bf16 [E][C][R]
// 32x32 LDS tile transpose, coalesced on both sides
// ============================================================
__global__ __launch_bounds__(256) void transpose_cvt_kernel(
    const float* __restrict__ src, __bf16* __restrict__ dst, int R, int C)
{
    __shared__ __bf16 tile[32][33];
    const size_t esz = (size_t)R * C;
    src += (size_t)blockIdx.z * esz;
    dst += (size_t)blockIdx.z * esz;
    const int cb = blockIdx.x * 32, rb = blockIdx.y * 32;
    const int c  = threadIdx.x & 31, r8 = threadIdx.x >> 5;
#pragma unroll
    for (int i = 0; i < 4; ++i) {
        int r = r8 + 8 * i;
        tile[r][c] = (__bf16)src[(size_t)(rb + r) * C + cb + c];
    }
    __syncthreads();
    const int rr = threadIdx.x & 31, c8 = threadIdx.x >> 5;
#pragma unroll
    for (int i = 0; i < 4; ++i) {
        int cc = c8 + 8 * i;
        dst[(size_t)(cb + cc) * R + rb + rr] = tile[rr][cc];
    }
}

// ============================================================
// Kernel 4: zero output
// ============================================================
__global__ __launch_bounds__(256) void zero_kernel(float4* __restrict__ out)
{
    out[blockIdx.x * 256 + threadIdx.x] = make_float4(0.f, 0.f, 0.f, 0.f);
}

// ------------------------------------------------------------
// fragment helpers: 16B-contiguous bf16 halves -> v16bf
// ------------------------------------------------------------
__device__ __forceinline__ v16bf frag_from(const __bf16* p) {
    v8bf lo = *(const v8bf*)(p);
    v8bf hi = *(const v8bf*)(p + 16);
    v16bf f;
#pragma unroll
    for (int j = 0; j < 8; ++j) { f[j] = lo[j]; f[j + 8] = hi[j]; }
    return f;
}

// ============================================================
// Kernel 5: fused SpikingSwiGLU expert per (16-token-tile, expert)
//  weights pre-transposed bf16: Wt[n][k] -> B frag = 2x b128 loads
// ============================================================
__global__ __launch_bounds__(256) void moe_kernel(
    const __bf16* __restrict__ xb,
    const float*  __restrict__ combine,
    const __bf16* __restrict__ Wgt,  const float* __restrict__ bgate,
    const __bf16* __restrict__ Wut,  const float* __restrict__ bup,
    const __bf16* __restrict__ Wdt,  const float* __restrict__ bdown,
    float* __restrict__ out)
{
    __shared__ __align__(16) __bf16 xs[T_TILE * XPAD];   // 33 KB
    __shared__ __align__(16) __bf16 hs[T_TILE * HPAD];   // 90 KB
    __shared__ float cs[T_TILE];

    const int e    = blockIdx.y;
    const int tb   = blockIdx.x * T_TILE;
    const int tid  = threadIdx.x;
    const int lane = tid & 31;
    const int wave = tid >> 5;
    const int m    = lane & 15;           // A row / B,C col within 16
    const int kb   = (lane >> 4) << 3;    // K sub-offset (0 or 8)
    const int rsh  = (lane >> 4) << 3;    // C row offset (0 or 8)

    if (tid < T_TILE) cs[tid] = combine[(size_t)(tb + tid) * N_EXP + e];
    {   // stage x tile: pure 16-byte bf16 copies
        const v8bf* xv = (const v8bf*)(xb + (size_t)tb * D_MODEL);
        for (int i = tid; i < T_TILE * D_MODEL / 8; i += 256) {
            int r = (i * 8) / D_MODEL, c = (i * 8) % D_MODEL;
            *(v8bf*)&xs[r * XPAD + c] = xv[i];
        }
    }
    __syncthreads();

    // per-row combine weights in registers (branchless epilogue)
    float cv[8];
#pragma unroll
    for (int i = 0; i < 8; ++i) cv[i] = cs[i + rsh];

    // ---------------- phase 1: gate & up ----------------
    const __bf16* Wg = Wgt + (size_t)e * D_MODEL * HIDDEN;   // [H][D]
    const __bf16* Wu = Wut + (size_t)e * D_MODEL * HIDDEN;

    for (int hc = wave * 16; hc < HIDDEN; hc += 8 * 16) {
        const int n = hc + m;
        const __bf16* wgp = Wg + (size_t)n * D_MODEL + kb;
        const __bf16* wup = Wu + (size_t)n * D_MODEL + kb;
        v8f accg = {};
        v8f accu = {};
        for (int k0 = 0; k0 < D_MODEL; k0 += 32) {
            v16bf a  = frag_from(&xs[m * XPAD + k0 + kb]);
            v16bf bg = frag_from(wgp + k0);
            v16bf bu = frag_from(wup + k0);
            accg = __builtin_amdgcn_wmma_f32_16x16x32_bf16(
                false, a, false, bg, (short)0, accg, false, false);
            accu = __builtin_amdgcn_wmma_f32_16x16x32_bf16(
                false, a, false, bu, (short)0, accu, false, false);
        }
        const float bgv = bgate[(size_t)e * HIDDEN + n];
        const float buv = bup[(size_t)e * HIDDEN + n];
#pragma unroll
        for (int i = 0; i < 8; ++i) {
            float g = accg[i] + bgv;
            float u = (accu[i] + buv) * cv[i];
            float h = (g > 1.0f) ? u : 0.0f;     // v_cndmask
            hs[(i + rsh) * HPAD + n] = (__bf16)h;
        }
    }
    __syncthreads();

    // ---------------- phase 2: down projection ----------------
    const __bf16* Wd = Wdt + (size_t)e * HIDDEN * D_MODEL;   // [D][H]

    for (int dc = wave * 16; dc < D_MODEL; dc += 8 * 16) {
        const int n = dc + m;
        const __bf16* wdp = Wd + (size_t)n * HIDDEN + kb;
        v8f acc = {};
        for (int k0 = 0; k0 < HIDDEN; k0 += 32) {
            v16bf a = frag_from(&hs[m * HPAD + k0 + kb]);
            v16bf b = frag_from(wdp + k0);
            acc = __builtin_amdgcn_wmma_f32_16x16x32_bf16(
                false, a, false, b, (short)0, acc, false, false);
        }
        const float bdv = bdown[(size_t)e * D_MODEL + n];
#pragma unroll
        for (int i = 0; i < 8; ++i) {
            float v = acc[i] + cv[i] * bdv;      // h pre-scaled by combine
            atomicAdd(&out[(size_t)(tb + i + rsh) * D_MODEL + n], v);
        }
    }
}

// ============================================================
extern "C" void kernel_launch(void* const* d_in, const int* in_sizes, int n_in,
                              void* d_out, int out_size, void* d_ws, size_t ws_size,
                              hipStream_t stream)
{
    const float* x      = (const float*)d_in[0];
    const float* Wg     = (const float*)d_in[1];
    const float* bg     = (const float*)d_in[2];
    const float* W_gate = (const float*)d_in[3];
    const float* b_gate = (const float*)d_in[4];
    const float* W_up   = (const float*)d_in[5];
    const float* b_up   = (const float*)d_in[6];
    const float* W_down = (const float*)d_in[7];
    const float* b_down = (const float*)d_in[8];
    float* out = (float*)d_out;

    // ---- workspace layout ----
    char* ws = (char*)d_ws;
    const size_t WSZ = (size_t)N_EXP * D_MODEL * HIDDEN;     // elems per weight tensor
    float*  combine = (float*)ws;                 ws += ((size_t)T_TOK * N_EXP * 4 + 255) & ~255ull;
    __bf16* xb      = (__bf16*)ws;                ws += ((size_t)T_TOK * D_MODEL * 2 + 255) & ~255ull;
    __bf16* Wgt     = (__bf16*)ws;                ws += (WSZ * 2 + 255) & ~255ull;
    __bf16* Wut     = (__bf16*)ws;                ws += (WSZ * 2 + 255) & ~255ull;
    __bf16* Wdt     = (__bf16*)ws;

    // 1) gating
    gate_kernel<<<T_TOK / 8, 256, 0, stream>>>(x, Wg, bg, combine);

    // 2) preprocess: x -> bf16; weights -> transposed bf16
    cvt_x_kernel<<<(T_TOK * D_MODEL) / (256 * 4), 256, 0, stream>>>((const float4*)x, xb);
    {
        dim3 g1(HIDDEN / 32, D_MODEL / 32, N_EXP);   // [D][H] -> [H][D]
        transpose_cvt_kernel<<<g1, 256, 0, stream>>>(W_gate, Wgt, D_MODEL, HIDDEN);
        transpose_cvt_kernel<<<g1, 256, 0, stream>>>(W_up,   Wut, D_MODEL, HIDDEN);
        dim3 g2(D_MODEL / 32, HIDDEN / 32, N_EXP);   // [H][D] -> [D][H]
        transpose_cvt_kernel<<<g2, 256, 0, stream>>>(W_down, Wdt, HIDDEN, D_MODEL);
    }

    // 3) zero output
    zero_kernel<<<(T_TOK * D_MODEL) / (256 * 4), 256, 0, stream>>>((float4*)out);

    // 4) fused experts
    dim3 grid(T_TOK / T_TILE, N_EXP);
    moe_kernel<<<grid, 256, 0, stream>>>(xb, combine,
                                         Wgt, b_gate, Wut, b_up,
                                         Wdt, b_down, out);
}
